// Alignment_17609365913938
// MI455X (gfx1250) — matrix operand order repaired and checked by reading, not compiled, not run on hardware
//
#include <hip/hip_runtime.h>
#include <hip/hip_bf16.h>

typedef __attribute__((ext_vector_type(4)))  _Float16 v4h;
typedef __attribute__((ext_vector_type(8)))  _Float16 v8h;
typedef __attribute__((ext_vector_type(16))) _Float16 v16h;
typedef __attribute__((ext_vector_type(8)))  float    v8f;

constexpr int kB = 32;
constexpr int kL = 1024;
constexpr int kH = 128;
constexpr int kRowsPerWG = 128;                 // 8 waves x 16 rows
constexpr int kWgPerBatch = kL / kRowsPerWG;    // 8
constexpr int kLdsStride = 40;                  // halves: 80B rows -> 16B aligned, conflict-friendly
constexpr float kLog2e = 1.4426950408889634f;
constexpr float kNeg2 = -10000.0f * 1.4426950408889634f;  // masked score in log2 domain

static __device__ __forceinline__ v16h join16(const _Float16* p0, const _Float16* p1) {
  v8h lo = *(const v8h*)p0;
  v8h hi = *(const v8h*)p1;
  v16h r;
#pragma unroll
  for (int i = 0; i < 8; ++i) { r[i] = lo[i]; r[i + 8] = hi[i]; }
  return r;
}

// A-matrix 16x32 f16 fragment: lane = (g<<4)|row ; element e -> K = 16g + e (contiguous)
static __device__ __forceinline__ v16h load_afrag(const _Float16* rowBase, int g) {
  return join16(rowBase + 16 * g, rowBase + 16 * g + 8);
}

// B-matrix 32x16 f16 fragment: lane = (g<<4)|col ; element e -> K = 16g + e (contiguous)
static __device__ __forceinline__ v16h load_bfrag(const _Float16* base) {
  return join16(base, base + 8);
}

static __device__ __forceinline__ v8f wmma16(v16h a, v16h b, v8f c) {
  return __builtin_amdgcn_wmma_f32_16x16x32_f16(false, a, false, b, (short)0, c, false, false);
}

static __device__ __forceinline__ float fexp2(float x) {
  return __builtin_amdgcn_exp2f(x);
}

// ---------------------------------------------------------------- f32 -> f16
__global__ void __launch_bounds__(256) k_cvt(const float* __restrict__ a,
                                             const float* __restrict__ b,
                                             _Float16* __restrict__ ah,
                                             _Float16* __restrict__ bh, int n4) {
  int i = blockIdx.x * blockDim.x + threadIdx.x;
  if (i >= n4) return;
  float4 va = ((const float4*)a)[i];
  float4 vb = ((const float4*)b)[i];
  v4h ha; ha[0] = (_Float16)va.x; ha[1] = (_Float16)va.y; ha[2] = (_Float16)va.z; ha[3] = (_Float16)va.w;
  v4h hb; hb[0] = (_Float16)vb.x; hb[1] = (_Float16)vb.y; hb[2] = (_Float16)vb.z; hb[3] = (_Float16)vb.w;
  ((v4h*)ah)[i] = ha;
  ((v4h*)bh)[i] = hb;
}

// ---------------------------------------------------- row softmax stats of X@Y^T
// All score math is in the log2 domain (temp pre-multiplied by log2(e)).
// Called twice: (a,b,mask_a,mask_b)->row stats of S ; (b,a,mask_b,mask_a)->col stats of S.
__global__ void __launch_bounds__(256) k_stats(const _Float16* __restrict__ X,
                                               const _Float16* __restrict__ Y,
                                               const int* __restrict__ mX,
                                               const int* __restrict__ mY,
                                               const float* __restrict__ tp,
                                               float* __restrict__ omax,
                                               float* __restrict__ osum) {
  const int wave = threadIdx.x >> 5;
  const int lane = threadIdx.x & 31;
  const int g    = lane >> 4;
  const int nr   = lane & 15;
  const int batch = blockIdx.x / kWgPerBatch;
  const int lBase = (blockIdx.x % kWgPerBatch) * kRowsPerWG + wave * 16;
  const float tl = (*tp) * kLog2e;              // temperature folded with log2(e)
  const _Float16* Xb = X + (size_t)batch * kL * kH;
  const _Float16* Yb = Y + (size_t)batch * kL * kH;

  v16h afrag[4];
  {
    const _Float16* arow = Xb + (size_t)(lBase + nr) * kH;
#pragma unroll
    for (int kc = 0; kc < 4; ++kc) afrag[kc] = load_afrag(arow + 32 * kc, g);
  }
  const int rbase = batch * kL + lBase + 8 * g;   // C-layout: this lane owns rows rbase..rbase+7
  int ma[8];
#pragma unroll
  for (int r = 0; r < 8; ++r) ma[r] = mX[rbase + r];

  float mx[8], sm[8];
#pragma unroll
  for (int r = 0; r < 8; ++r) { mx[r] = -3.0e38f; sm[r] = 0.0f; }

  for (int mt = 0; mt < kL / 16; ++mt) {
    const int m = mt * 16 + nr;                   // this lane's column
    const _Float16* yrow = Yb + (size_t)m * kH;
    const int mb = mY[batch * kL + m];
    v16h bf[4];
#pragma unroll
    for (int kc = 0; kc < 4; ++kc) bf[kc] = load_bfrag(yrow + 32 * kc + 16 * g);
    v8f acc = {};
#pragma unroll
    for (int kc = 0; kc < 4; ++kc) acc = wmma16(afrag[kc], bf[kc], acc);
#pragma unroll
    for (int r = 0; r < 8; ++r) {
      float s = (ma[r] != 0 && mb != 0) ? acc[r] * tl : kNeg2;
      float nm = fmaxf(mx[r], s);
      sm[r] = sm[r] * fexp2(mx[r] - nm) + fexp2(s - nm);
      mx[r] = nm;
    }
  }
  // combine 16 lanes of each half-wave (online-softmax merge)
#pragma unroll
  for (int off = 1; off < 16; off <<= 1) {
#pragma unroll
    for (int r = 0; r < 8; ++r) {
      float omx = __shfl_xor(mx[r], off, 32);
      float osm = __shfl_xor(sm[r], off, 32);
      float nm = fmaxf(mx[r], omx);
      sm[r] = sm[r] * fexp2(mx[r] - nm) + osm * fexp2(omx - nm);
      mx[r] = nm;
    }
  }
  if (nr == 0) {
#pragma unroll
    for (int r = 0; r < 8; ++r) { omax[rbase + r] = mx[r]; osum[rbase + r] = sm[r]; }
  }
}

// --------------------------------------------------------------- fused features
__global__ void __launch_bounds__(256) k_fused(const _Float16* __restrict__ ah,
                                               const _Float16* __restrict__ bh,
                                               const int* __restrict__ mA,
                                               const int* __restrict__ mB,
                                               const float* __restrict__ tp,
                                               const float* __restrict__ rmax,
                                               const float* __restrict__ rsum,
                                               const float* __restrict__ cmax,
                                               const float* __restrict__ csum,
                                               float* __restrict__ fa,
                                               float* __restrict__ fb) {
  __shared__ _Float16 sA[kH * kLdsStride];              // a chunk transposed [d][m]
  __shared__ _Float16 sB[kH * kLdsStride];              // b chunk transposed [d][m]
  __shared__ _Float16 sPQ[8 * 2 * 16 * kLdsStride];     // per-wave P,Q tiles [16][40]

  const int wave = threadIdx.x >> 5;
  const int lane = threadIdx.x & 31;
  const int g    = lane >> 4;
  const int nr   = lane & 15;
  const int batch = blockIdx.x / kWgPerBatch;
  const int lBase = (blockIdx.x % kWgPerBatch) * kRowsPerWG + wave * 16;
  const float tl = (*tp) * kLog2e;
  const _Float16* Ab = ah + (size_t)batch * kL * kH;
  const _Float16* Bb = bh + (size_t)batch * kL * kH;

  v16h afrag[4];
  {
    const _Float16* arow = Ab + (size_t)(lBase + nr) * kH;
#pragma unroll
    for (int kc = 0; kc < 4; ++kc) afrag[kc] = load_afrag(arow + 32 * kc, g);
  }
  const int rbase = batch * kL + lBase + 8 * g;
  int ma[8]; float rmx[8], rin[8];
#pragma unroll
  for (int r = 0; r < 8; ++r) {
    ma[r]  = mA[rbase + r];
    rmx[r] = rmax[rbase + r];
    rin[r] = 1.0f / rsum[rbase + r];
  }

  v8f facc[8] = {};
  v8f gacc[8] = {};

  _Float16* Pl = sPQ + wave * (2 * 16 * kLdsStride);
  _Float16* Ql = Pl + 16 * kLdsStride;

  const int tsm = threadIdx.x >> 3;         // 0..31 : m within chunk
  const int tsd = (threadIdx.x & 7) * 16;   // 0..112: d group

  for (int m0 = 0; m0 < kL; m0 += 32) {
    // stage a/b chunk [32 x 128] transposed into LDS (all 8 waves cooperate)
    {
      const _Float16* pa = Ab + (size_t)(m0 + tsm) * kH + tsd;
      const _Float16* pb = Bb + (size_t)(m0 + tsm) * kH + tsd;
      v8h a0 = *(const v8h*)pa; v8h a1 = *(const v8h*)(pa + 8);
      v8h b0 = *(const v8h*)pb; v8h b1 = *(const v8h*)(pb + 8);
#pragma unroll
      for (int i = 0; i < 8; ++i) {
        sA[(tsd + i) * kLdsStride + tsm]     = a0[i];
        sA[(tsd + 8 + i) * kLdsStride + tsm] = a1[i];
        sB[(tsd + i) * kLdsStride + tsm]     = b0[i];
        sB[(tsd + 8 + i) * kLdsStride + tsm] = b1[i];
      }
      if (m0 + 32 < kL) {
        __builtin_prefetch(pa + 32 * kH, 0, 0);
        __builtin_prefetch(pb + 32 * kH, 0, 0);
      }
    }
    __syncthreads();

    // scores for two 16-wide m tiles; P (row-softmax) and Q (col-softmax) -> LDS
#pragma unroll
    for (int tt = 0; tt < 2; ++tt) {
      const int m = m0 + tt * 16 + nr;
      const _Float16* yrow = Bb + (size_t)m * kH;
      const int mb = mB[batch * kL + m];
      const float cm = cmax[batch * kL + m];
      const float ci = 1.0f / csum[batch * kL + m];
      v16h bf[4];
#pragma unroll
      for (int kc = 0; kc < 4; ++kc) bf[kc] = load_bfrag(yrow + 32 * kc + 16 * g);
      v8f acc = {};
#pragma unroll
      for (int kc = 0; kc < 4; ++kc) acc = wmma16(afrag[kc], bf[kc], acc);
#pragma unroll
      for (int r = 0; r < 8; ++r) {
        float s = (ma[r] != 0 && mb != 0) ? acc[r] * tl : kNeg2;
        float p = fexp2(s - rmx[r]) * rin[r];
        float q = fexp2(s - cm) * ci;
        Pl[(8 * g + r) * kLdsStride + tt * 16 + nr] = (_Float16)p;
        Ql[(8 * g + r) * kLdsStride + tt * 16 + nr] = (_Float16)q;
      }
    }
    // wave-private LDS round-trip: ensure stores landed before A-frag reads
    asm volatile("s_wait_dscnt 0x0" ::: "memory");

    const v16h pf = load_afrag(Pl + nr * kLdsStride, g);
    const v16h qf = load_afrag(Ql + nr * kLdsStride, g);

    // feature GEMMs with double-buffered LDS B-fragments
    v16h bfB = load_bfrag(&sB[nr * kLdsStride + 16 * g]);
    v16h bfA = load_bfrag(&sA[nr * kLdsStride + 16 * g]);
#pragma unroll
    for (int nc = 0; nc < 8; ++nc) {
      v16h nB = bfB, nA = bfA;
      if (nc < 7) {
        nB = load_bfrag(&sB[(16 * (nc + 1) + nr) * kLdsStride + 16 * g]);
        nA = load_bfrag(&sA[(16 * (nc + 1) + nr) * kLdsStride + 16 * g]);
      }
      facc[nc] = wmma16(pf, bfB, facc[nc]);               // feature_a += P @ b
      gacc[nc] = wmma16(qf, bfA, gacc[nc]);               // feature_b += Q @ a
      bfB = nB; bfA = nA;
    }
    __syncthreads();
  }

#pragma unroll
  for (int nc = 0; nc < 8; ++nc) {
#pragma unroll
    for (int r = 0; r < 8; ++r) {
      const int row = lBase + 8 * g + r;
      const size_t o = ((size_t)batch * kL + row) * kH + 16 * nc + nr;
      fa[o] = facc[nc][r];
      fb[o] = gacc[nc][r];
    }
  }
}

// ------------------------------------------------------------------- launcher
extern "C" void kernel_launch(void* const* d_in, const int* in_sizes, int n_in,
                              void* d_out, int out_size, void* d_ws, size_t ws_size,
                              hipStream_t stream) {
  (void)in_sizes; (void)n_in; (void)out_size; (void)ws_size;
  const float* a      = (const float*)d_in[0];
  const float* b      = (const float*)d_in[1];
  const int*   mask_a = (const int*)d_in[2];
  const int*   mask_b = (const int*)d_in[3];
  const float* temp   = (const float*)d_in[4];

  const size_t BLH = (size_t)kB * kL * kH;
  _Float16* ah = (_Float16*)d_ws;
  _Float16* bh = ah + BLH;
  float* rmx = (float*)(bh + BLH);
  float* rsm = rmx + (size_t)kB * kL;
  float* cmx = rsm + (size_t)kB * kL;
  float* csm = cmx + (size_t)kB * kL;

  float* fa = (float*)d_out;
  float* fb = fa + BLH;

  const int n4 = (int)(BLH / 4);
  hipLaunchKernelGGL(k_cvt, dim3(n4 / 256), dim3(256), 0, stream, a, b, ah, bh, n4);

  dim3 grid(kB * kWgPerBatch), blk(256);
  // row stats of S  (softmax over axis=2 -> attention_b)
  hipLaunchKernelGGL(k_stats, grid, blk, 0, stream, ah, bh, mask_a, mask_b, temp, rmx, rsm);
  // col stats of S = row stats of S^T (softmax over axis=1 -> attention_a)
  hipLaunchKernelGGL(k_stats, grid, blk, 0, stream, bh, ah, mask_b, mask_a, temp, cmx, csm);
  hipLaunchKernelGGL(k_fused, grid, blk, 0, stream, ah, bh, mask_a, mask_b, temp,
                     rmx, rsm, cmx, csm, fa, fb);
}
